// HeteroGNN_68642167324726
// MI455X (gfx1250) — matrix operand order repaired and checked by reading, not compiled, not run on hardware
//
#include <hip/hip_runtime.h>
#include <hip/hip_bf16.h>

#define HDIM 512
#define NEG_SLOPE 0.01f
#define ASTRIDE 1032  // LDS row stride in halves: 2064B = 16B-aligned, bank-skewed

typedef __attribute__((ext_vector_type(16))) __bf16 v16bf;
typedef __attribute__((ext_vector_type(8)))  __bf16 v8bf;
typedef __attribute__((ext_vector_type(4)))  __bf16 v4bf;
typedef __attribute__((ext_vector_type(8)))  float  v8f;

// ---------------------------------------------------------------------------
// A-fragment (16x32 bf16, MxK) from an f32 row with fused scale (mean = msg/cnt).
// Lane l holds row m = l&15; halves 0..7 -> K = kbase + hi*8 + i,
// halves 8..15 -> K = kbase + 16 + hi*8 + (i-8).
// ---------------------------------------------------------------------------
static __device__ __forceinline__ v16bf frag_a_f32(const float* __restrict__ row,
                                                   int kbase, int hi, float scale) {
  const float* p0 = row + kbase + hi * 8;
  const float* p1 = p0 + 16;
  float4 a0 = *(const float4*)(p0 + 0);
  float4 a1 = *(const float4*)(p0 + 4);
  float4 b0 = *(const float4*)(p1 + 0);
  float4 b1 = *(const float4*)(p1 + 4);
  v16bf r;
  r[0]  = (__bf16)(a0.x * scale); r[1]  = (__bf16)(a0.y * scale);
  r[2]  = (__bf16)(a0.z * scale); r[3]  = (__bf16)(a0.w * scale);
  r[4]  = (__bf16)(a1.x * scale); r[5]  = (__bf16)(a1.y * scale);
  r[6]  = (__bf16)(a1.z * scale); r[7]  = (__bf16)(a1.w * scale);
  r[8]  = (__bf16)(b0.x * scale); r[9]  = (__bf16)(b0.y * scale);
  r[10] = (__bf16)(b0.z * scale); r[11] = (__bf16)(b0.w * scale);
  r[12] = (__bf16)(b1.x * scale); r[13] = (__bf16)(b1.y * scale);
  r[14] = (__bf16)(b1.z * scale); r[15] = (__bf16)(b1.w * scale);
  return r;
}

// A-fragment from a bf16 row (global or LDS): two contiguous 16-byte runs.
static __device__ __forceinline__ v16bf frag_a_bf(const __bf16* __restrict__ row,
                                                  int kbase, int hi) {
  v8bf lo = *(const v8bf*)(row + kbase + hi * 8);
  v8bf up = *(const v8bf*)(row + kbase + 16 + hi * 8);
  return __builtin_shufflevector(lo, up, 0, 1, 2, 3, 4, 5, 6, 7,
                                 8, 9, 10, 11, 12, 13, 14, 15);
}

// B-fragment (32x16 bf16, KxN): lane l holds column n = l&15; halves 0..7 ->
// K = kbase + hi*8 + i, halves 8..15 -> K = kbase + 16 + hi*8 + (i-8) (B[k][n]=W[n][k]).
static __device__ __forceinline__ v16bf frag_b_bf(const __bf16* __restrict__ wrow,
                                                  int kbase, int hi) {
  v8bf a = *(const v8bf*)(wrow + kbase + hi * 8);
  v8bf b = *(const v8bf*)(wrow + kbase + 16 + hi * 8);
  return __builtin_shufflevector(a, b, 0, 1, 2, 3, 4, 5, 6, 7,
                                 8, 9, 10, 11, 12, 13, 14, 15);
}

// Per-lane async global->LDS 16-byte copy (CDNA5, tracked by ASYNCcnt).
static __device__ __forceinline__ void async_g2l_b128(unsigned lds_byte_addr,
                                                      const void* gaddr) {
  asm volatile("global_load_async_to_lds_b128 %0, %1, off"
               :: "v"(lds_byte_addr), "v"(gaddr) : "memory");
}

static __device__ __forceinline__ void wait_asynccnt0() {
  asm volatile("s_wait_asynccnt 0x0" ::: "memory");
}

// ---------------------------------------------------------------------------
// f32 -> bf16 bulk conversion (8 elements / thread).
// ---------------------------------------------------------------------------
__global__ __launch_bounds__(256) void cvt_bf16_kernel(
    const float* __restrict__ in, __bf16* __restrict__ out, long long n) {
  long long i = ((long long)blockIdx.x * blockDim.x + threadIdx.x) * 8;
  if (i + 8 <= n) {
    float4 a = *(const float4*)(in + i);
    float4 b = *(const float4*)(in + i + 4);
    v8bf o;
    o[0] = (__bf16)a.x; o[1] = (__bf16)a.y; o[2] = (__bf16)a.z; o[3] = (__bf16)a.w;
    o[4] = (__bf16)b.x; o[5] = (__bf16)b.y; o[6] = (__bf16)b.z; o[7] = (__bf16)b.w;
    *(v8bf*)(out + i) = o;
  } else {
    for (long long j = i; j < n; ++j) out[j] = (__bf16)in[j];
  }
}

// ---------------------------------------------------------------------------
// Segment-sum scatter: msg[dst] += x[src] (bf16 source, f32 atomic accum).
// ---------------------------------------------------------------------------
__global__ __launch_bounds__(256) void scatter_add_kernel(
    const __bf16* __restrict__ xsrc, const int* __restrict__ src,
    const int* __restrict__ dst, float* __restrict__ msg,
    float* __restrict__ cnt, int E) {
  long long tid = (long long)blockIdx.x * blockDim.x + threadIdx.x;
  if (tid >= (long long)E * (HDIM / 4)) return;
  int e = (int)(tid >> 7);
  int c = (int)(tid & 127);
  int s = src[e], d = dst[e];
  v4bf v = *(const v4bf*)(xsrc + (size_t)s * HDIM + c * 4);
  float* mp = msg + (size_t)d * HDIM + c * 4;
  atomicAdd(mp + 0, (float)v[0]);
  atomicAdd(mp + 1, (float)v[1]);
  atomicAdd(mp + 2, (float)v[2]);
  atomicAdd(mp + 3, (float)v[3]);
  if (c == 0) atomicAdd(cnt + d, 1.0f);
}

// ---------------------------------------------------------------------------
// Fused SAGE layer: y = leaky_relu((msg/cnt) @ Wagg^T + x @ Wself^T + bias),
// stored as bf16.  256 threads = 8 waves; block covers 32 rows x 512 cols.
// ---------------------------------------------------------------------------
__global__ __launch_bounds__(256) void sage_gemm_kernel(
    const float* __restrict__ msg, const float* __restrict__ cnt,
    const __bf16* __restrict__ xdst,
    const __bf16* __restrict__ Wagg, const __bf16* __restrict__ Wself,
    const float* __restrict__ bias, __bf16* __restrict__ y, int N) {
  int lane = threadIdx.x & 31;
  int wave = threadIdx.x >> 5;
  int hi = lane >> 4;
  int lme = lane & 15;
  int mrow0 = blockIdx.x * 32 + (wave & 1) * 16;
  int g = wave >> 1;  // column group: cols [g*128, g*128+128)

  int arow = mrow0 + lme;
  if (arow >= N) arow = N - 1;
  const float*  msgRow = msg + (size_t)arow * HDIM;
  const __bf16* xRow   = xdst + (size_t)arow * HDIM;
  float scale = 1.0f / fmaxf(cnt[arow], 1.0f);

  v8f vzero = {};
  v8f acc[8];
#pragma unroll
  for (int t = 0; t < 8; ++t) acc[t] = vzero;

  for (int kc = 0; kc < HDIM; kc += 32) {
    __builtin_prefetch(msgRow + kc + 64, 0, 1);  // global_prefetch_b8
    __builtin_prefetch(xRow + kc + 128, 0, 1);
    v16bf aM = frag_a_f32(msgRow, kc, hi, scale);
    v16bf aX = frag_a_bf(xRow, kc, hi);
#pragma unroll
    for (int t = 0; t < 8; ++t) {
      int n = g * 128 + t * 16 + lme;
      v16bf bL = frag_b_bf(Wagg + (size_t)n * HDIM, kc, hi);
      acc[t] = __builtin_amdgcn_wmma_f32_16x16x32_bf16(
          false, aM, false, bL, (short)0, acc[t], false, false);
      v16bf bR = frag_b_bf(Wself + (size_t)n * HDIM, kc, hi);
      acc[t] = __builtin_amdgcn_wmma_f32_16x16x32_bf16(
          false, aX, false, bR, (short)0, acc[t], false, false);
    }
  }

  // C/D layout: lane holds (m = r + 8*hi, n = lane&15) in VGPR r.
#pragma unroll
  for (int t = 0; t < 8; ++t) {
    int n = g * 128 + t * 16 + lme;
    float bn = bias[n];
#pragma unroll
    for (int r = 0; r < 8; ++r) {
      int m = mrow0 + r + 8 * hi;
      if (m < N) {
        float v = acc[t][r] + bn;
        v = (v > 0.0f) ? v : v * NEG_SLOPE;
        y[(size_t)m * HDIM + n] = (__bf16)v;
      }
    }
  }
}

// ---------------------------------------------------------------------------
// Edge MLP: out[e,:] = relu([xG[row], xD[col]] @ W1^T + b1) @ W2^T + b2.
// Block = 16 edges.  The 16x1024 bf16 A-panel (concat of xG[row], xD[col]) is
// staged ONCE per block into LDS with GLOBAL_LOAD_ASYNC_TO_LDS_B128 (async DMA,
// ASYNCcnt) -- removes the 8x redundant per-wave gather of the same edges.
// 8 waves each own 64 cols of h (4 N-tiles); K = 1024 from LDS.
// lin2 fused via LDS ds_add_f32 reduction; out written once, non-atomically.
// ---------------------------------------------------------------------------
__global__ __launch_bounds__(256) void edge_mlp_kernel(
    const __bf16* __restrict__ xG, const __bf16* __restrict__ xD,
    const int* __restrict__ eSrc, const int* __restrict__ eDst,
    const __bf16* __restrict__ W1, const float* __restrict__ b1,
    const float* __restrict__ W2, const float* __restrict__ b2,
    float* __restrict__ out, int E) {
  __shared__ __bf16 Atile[16 * ASTRIDE];  // 33 KB
  __shared__ float red[16][2];

  int tid = threadIdx.x;
  int lane = tid & 31;
  int wave = tid >> 5;  // col group of 64
  int hi = lane >> 4;
  int lme = lane & 15;

  if (tid < 32) red[tid >> 1][tid & 1] = 0.0f;

  int e0 = blockIdx.x * 16;

  // ---- async stage: 16 rows x 1024 halves (2KB/row) into LDS ----
  {
    int r = tid >> 4;            // edge row 0..15
    int c0h = (tid & 15) * 64;   // this thread covers halves [c0h, c0h+64)
    int e = e0 + r;
    if (e >= E) e = E - 1;
    const __bf16* gsrc = (c0h < HDIM)
                             ? (xG + (size_t)eSrc[e] * HDIM + c0h)
                             : (xD + (size_t)eDst[e] * HDIM + (c0h - HDIM));
    unsigned ldst = (unsigned)(size_t)&Atile[r * ASTRIDE + c0h];
#pragma unroll
    for (int i = 0; i < 8; ++i) {
      async_g2l_b128(ldst + i * 16, gsrc + i * 8);
    }
  }
  wait_asynccnt0();
  __syncthreads();

  v8f vzero = {};
  v8f acc[4];
#pragma unroll
  for (int t = 0; t < 4; ++t) acc[t] = vzero;

  const __bf16* aRow = &Atile[lme * ASTRIDE];
  for (int kc = 0; kc < 2 * HDIM; kc += 32) {
    v16bf a = frag_a_bf(aRow, kc, hi);  // ds_load_b128 x2
#pragma unroll
    for (int t = 0; t < 4; ++t) {
      int n = wave * 64 + t * 16 + lme;
      v16bf bb = frag_b_bf(W1 + (size_t)n * (2 * HDIM), kc, hi);
      acc[t] = __builtin_amdgcn_wmma_f32_16x16x32_bf16(
          false, a, false, bb, (short)0, acc[t], false, false);
    }
  }

#pragma unroll
  for (int r = 0; r < 8; ++r) {
    int m = r + 8 * hi;
    float s0 = 0.0f, s1 = 0.0f;
#pragma unroll
    for (int t = 0; t < 4; ++t) {
      int n = wave * 64 + t * 16 + lme;
      float h = acc[t][r] + b1[n];
      h = fmaxf(h, 0.0f);
      s0 += h * W2[n];
      s1 += h * W2[HDIM + n];
    }
    atomicAdd(&red[m][0], s0);  // ds_add_f32
    atomicAdd(&red[m][1], s1);
  }
  __syncthreads();

  if (tid < 32) {
    int m = tid >> 1;
    int j = tid & 1;
    int e = e0 + m;
    if (e < E) out[(size_t)e * 2 + j] = red[m][j] + b2[j];
  }
}

// ---------------------------------------------------------------------------
// Host orchestration (graph-capture safe: async memset + launches only).
// ---------------------------------------------------------------------------
extern "C" void kernel_launch(void* const* d_in, const int* in_sizes, int n_in,
                              void* d_out, int out_size, void* d_ws, size_t ws_size,
                              hipStream_t stream) {
  const float* x_G    = (const float*)d_in[0];
  const float* x_D    = (const float*)d_in[1];
  const int*   eGD    = (const int*)d_in[2];   // [2,E]: row0 src(G), row1 dst(D)
  const int*   eDG    = (const int*)d_in[3];   // [2,E]: row0 src(D), row1 dst(G)
  const float* Wl     = (const float*)d_in[4]; // [2,2,H,H]
  const float* Wr     = (const float*)d_in[5]; // [2,2,H,H]
  const float* bvec   = (const float*)d_in[6]; // [2,2,H]
  const float* lin1_w = (const float*)d_in[7]; // [H, 2H]
  const float* lin1_b = (const float*)d_in[8];
  const float* lin2_w = (const float*)d_in[9]; // [2, H]
  const float* lin2_b = (const float*)d_in[10];

  int NG = in_sizes[0] / HDIM;
  int ND = in_sizes[1] / HDIM;
  int E  = in_sizes[2] / 2;
  long long nWl   = in_sizes[4];  // 2*2*H*H
  long long nWr   = in_sizes[5];
  long long nLin1 = in_sizes[7];  // H*2H

  // f32 region: atomic accumulation buffers + counts
  float* f = (float*)d_ws;
  float* msgG = f; f += (size_t)NG * HDIM;
  float* msgD = f; f += (size_t)ND * HDIM;
  float* cntG = f; f += ((size_t)NG + 7) & ~(size_t)7;
  float* cntD = f; f += ((size_t)ND + 7) & ~(size_t)7;

  // bf16 region: activations (ping-pong) + converted weights
  __bf16* h = (__bf16*)f;
  __bf16* xGa  = h; h += (size_t)NG * HDIM;
  __bf16* xGb  = h; h += (size_t)NG * HDIM;
  __bf16* xDa  = h; h += (size_t)ND * HDIM;
  __bf16* xDb  = h; h += (size_t)ND * HDIM;
  __bf16* Wl16 = h; h += (size_t)nWl;
  __bf16* Wr16 = h; h += (size_t)nWr;
  __bf16* L116 = h; h += (size_t)nLin1;

  dim3 blk(256);
  // One-time (per launch) f32 -> bf16 conversions: inputs + all weights.
  long long nXG = (long long)NG * HDIM, nXD = (long long)ND * HDIM;
  cvt_bf16_kernel<<<(int)((nXG / 8 + 255) / 256), blk, 0, stream>>>(x_G, xGa, nXG);
  cvt_bf16_kernel<<<(int)((nXD / 8 + 255) / 256), blk, 0, stream>>>(x_D, xDa, nXD);
  cvt_bf16_kernel<<<(int)((nWl / 8 + 255) / 256), blk, 0, stream>>>(Wl, Wl16, nWl);
  cvt_bf16_kernel<<<(int)((nWr / 8 + 255) / 256), blk, 0, stream>>>(Wr, Wr16, nWr);
  cvt_bf16_kernel<<<(int)((nLin1 / 8 + 255) / 256), blk, 0, stream>>>(lin1_w, L116, nLin1);

  long long scatThreads = (long long)E * (HDIM / 4);
  int scatBlocks = (int)((scatThreads + 255) / 256);

  for (int l = 0; l < 2; ++l) {
    // G -> D (edge type 0, updates D)
    hipMemsetAsync(msgD, 0, (size_t)ND * HDIM * sizeof(float), stream);
    hipMemsetAsync(cntD, 0, (size_t)ND * sizeof(float), stream);
    scatter_add_kernel<<<scatBlocks, blk, 0, stream>>>(xGa, eGD, eGD + E,
                                                       msgD, cntD, E);
    // D -> G (edge type 1, updates G)
    hipMemsetAsync(msgG, 0, (size_t)NG * HDIM * sizeof(float), stream);
    hipMemsetAsync(cntG, 0, (size_t)NG * sizeof(float), stream);
    scatter_add_kernel<<<scatBlocks, blk, 0, stream>>>(xDa, eDG, eDG + E,
                                                       msgG, cntG, E);

    size_t w0 = ((size_t)l * 2 + 0) * HDIM * HDIM;
    size_t w1 = ((size_t)l * 2 + 1) * HDIM * HDIM;
    size_t b0 = ((size_t)l * 2 + 0) * HDIM;
    size_t b1 = ((size_t)l * 2 + 1) * HDIM;

    sage_gemm_kernel<<<(ND + 31) / 32, blk, 0, stream>>>(
        msgD, cntD, xDa, Wl16 + w0, Wr16 + w0, bvec + b0, xDb, ND);
    sage_gemm_kernel<<<(NG + 31) / 32, blk, 0, stream>>>(
        msgG, cntG, xGa, Wl16 + w1, Wr16 + w1, bvec + b1, xGb, NG);

    __bf16* t;
    t = xGa; xGa = xGb; xGb = t;
    t = xDa; xDa = xDb; xDb = t;
  }

  edge_mlp_kernel<<<(E + 15) / 16, blk, 0, stream>>>(
      xGa, xDa, eGD, eGD + E, L116, lin1_b, lin2_w, lin2_b,
      (float*)d_out, E);
}
